// XorzoTransformer_59287728554108
// MI455X (gfx1250) — matrix operand order repaired and checked by reading, not compiled, not run on hardware
//
#include <hip/hip_runtime.h>
#include <hip/hip_bf16.h>

// ---------------------------------------------------------------------------
// CDNA5 / gfx1250 WMMA transformer block.
// bf16 WMMA (v_wmma_f32_16x16x32_bf16) for all GEMMs + flash attention.
// Weights pre-transposed to N-major; all inner-loop fragment loads are
// paired ds_load_b128. Register-staged, double-buffered LDS pipelines.
// 128-thread blocks (4 waves): in CU mode this doubles the per-wave VGPR
// budget so staging registers are NOT spilled to scratch.
// ---------------------------------------------------------------------------

typedef __attribute__((ext_vector_type(16))) __bf16 v16bf;
typedef __attribute__((ext_vector_type(8)))  float  v8f;

#define D_      1024
#define T_      1024
#define B_      4
#define H_      16
#define DH_     64
#define BT_     4096
#define DHID_   2108
#define DHIDP_  2176      // padded to multiple of 64 (GEMM BN)
#define DACT_   1656
#define INV_SCALE_ 0.125f // 1/sqrt(DH), layer 0
#define PI_     3.14159265358979f

__device__ __forceinline__ float sigm(float x) { return 1.f / (1.f + __expf(-x)); }
__device__ __forceinline__ float gelu_exact(float x) {
  return 0.5f * x * (1.f + erff(x * 0.70710678118654752f));
}

#define WMMA_BF16(A, Bv, C) \
  __builtin_amdgcn_wmma_f32_16x16x32_bf16(false, (A), false, (Bv), (short)0, (C), false, false)

// A fragment (16x32, MxK) from M-major storage: lane&15 = row M; the 16
// elements are one contiguous 16-half (32B) run at K = hi*16.
__device__ __forceinline__ v16bf frag_a_vec(const __bf16* base, int stride) {
  int lane = threadIdx.x & 31, lh = lane & 15, hi = lane >> 4;
  const __bf16* p0 = base + lh * stride + hi * 16;
  union { v16bf v; uint4 u[2]; } U;
  U.u[0] = *(const uint4*)(p0);
  U.u[1] = *(const uint4*)(p0 + 8);
  return U.v;
}

// B fragment (32x16, KxN) from N-major storage base[n*stride + k]:
// lane&15 = col N; the 16 elements are one contiguous 16-half (32B) run
// at K = hi*16.  Two aligned b128 loads.
__device__ __forceinline__ v16bf frag_b_vec(const __bf16* base, int stride) {
  int lane = threadIdx.x & 31, lh = lane & 15, hi = lane >> 4;
  const __bf16* p0 = base + lh * stride + hi * 16;
  union { v16bf v; uint4 u[2]; } U;
  U.u[0] = *(const uint4*)(p0);
  U.u[1] = *(const uint4*)(p0 + 8);
  return U.v;
}

// ---------------------------------------------------------------------------
// Generic bf16 GEMM: C = A(MxK) * B(KxN) with B supplied TRANSPOSED
// (Bt is N rows x K cols, row-major). f32 accumulate, fused epilogues.
// Block = 128 threads = 4 waves. Tile 128x64, wave tile 32x64, K-slab 32.
// Named-scalar register staging + double-buffered LDS: 1 barrier per slab,
// next slab's global loads in flight during the WMMA burst.
// ---------------------------------------------------------------------------
#define GBM 128
#define GBN 64
#define GBK 32
#define TSTR 40    // 32 + 8 pad halfs; 80B rows -> 16B aligned

__global__ __launch_bounds__(128, 1) void gemm_bf16_wmma(
    const __bf16* __restrict__ A, const __bf16* __restrict__ Bt,
    int M, int N, int K,
    __bf16* __restrict__ outB, float* __restrict__ outF,
    const float* __restrict__ bias, int biasN,
    const float* __restrict__ res, const float* __restrict__ rbPtr,
    int nActive, int mode)
{
  __shared__ __bf16 As[2][GBM * TSTR];
  __shared__ __bf16 Bs[2][GBN * TSTR];
  int tid  = threadIdx.x;
  int wave = tid >> 5;
  int lane = tid & 31, lh = lane & 15, hi = lane >> 4;
  int m0 = blockIdx.y * GBM;
  int n0 = blockIdx.x * GBN;
  int wm = wave * 32;

  // staging geometry: chunk cc covers rows (tid>>2)+32*cc, cols (tid&3)*8
  int r0 = tid >> 2, q8 = (tid & 3) * 8;
  const __bf16* pA = A  + (size_t)(m0 + r0) * K + q8;
  const __bf16* pB = Bt + (size_t)(n0 + r0) * K + q8;
  size_t st32 = (size_t)32 * K;
  int ldsO = r0 * TSTR + q8;

  v8f acc[2][4] = {};
  uint4 ta0, ta1, ta2, ta3, tb0, tb1;

  // stage first slab into registers
  ta0 = *(const uint4*)(pA);
  ta1 = *(const uint4*)(pA + st32);
  ta2 = *(const uint4*)(pA + 2 * st32);
  ta3 = *(const uint4*)(pA + 3 * st32);
  tb0 = *(const uint4*)(pB);
  tb1 = *(const uint4*)(pB + st32);
  pA += GBK; pB += GBK;

  int cur = 0;
  for (int k0 = 0; k0 < K; k0 += GBK) {
    // commit staged registers to LDS[cur]
    *(uint4*)&As[cur][ldsO]             = ta0;
    *(uint4*)&As[cur][ldsO + 32 * TSTR] = ta1;
    *(uint4*)&As[cur][ldsO + 64 * TSTR] = ta2;
    *(uint4*)&As[cur][ldsO + 96 * TSTR] = ta3;
    *(uint4*)&Bs[cur][ldsO]             = tb0;
    *(uint4*)&Bs[cur][ldsO + 32 * TSTR] = tb1;
    __syncthreads();

    if (k0 + GBK < K) {                        // issue next slab's loads now
      ta0 = *(const uint4*)(pA);
      ta1 = *(const uint4*)(pA + st32);
      ta2 = *(const uint4*)(pA + 2 * st32);
      ta3 = *(const uint4*)(pA + 3 * st32);
      tb0 = *(const uint4*)(pB);
      tb1 = *(const uint4*)(pB + st32);
      __builtin_prefetch(pA + GBK, 0, 0);      // warm L2 two slabs ahead
      __builtin_prefetch(pB + GBK, 0, 0);
      pA += GBK; pB += GBK;
    }

    v16bf a0 = frag_a_vec(&As[cur][(wm)      * TSTR], TSTR);
    v16bf a1 = frag_a_vec(&As[cur][(wm + 16) * TSTR], TSTR);
#pragma unroll
    for (int j = 0; j < 4; ++j) {
      v16bf bf = frag_b_vec(&Bs[cur][(j * 16) * TSTR], TSTR);
      acc[0][j] = WMMA_BF16(a0, bf, acc[0][j]);
      acc[1][j] = WMMA_BF16(a1, bf, acc[1][j]);
    }
    cur ^= 1;
  }

  float rb = rbPtr ? sigm(rbPtr[0]) : 0.f;

#pragma unroll
  for (int i = 0; i < 2; ++i)
#pragma unroll
    for (int j = 0; j < 4; ++j)
#pragma unroll
      for (int r = 0; r < 8; ++r) {
        int row = m0 + wm + i * 16 + hi * 8 + r;
        int col = n0 + j * 16 + lh;
        size_t idx = (size_t)row * N + col;
        float v = acc[i][j][r];
        if (mode == 0) {                       // bf16 store
          outB[idx] = (__bf16)v;
        } else if (mode == 1) {                // residual f32
          outF[idx] = res[idx] + rb * v;
        } else if (mode == 2) {                // bias + active-mask + gelu -> bf16
          float t = v + (col < biasN ? bias[col] : 0.f);
          if (col >= nActive) t = 0.f;
          outB[idx] = (__bf16)gelu_exact(t);
        } else if (mode == 3) {                // bias + residual -> f32 + bf16
          float t = v + (col < biasN ? bias[col] : 0.f);
          float y = res[idx] + rb * t;
          outF[idx] = y;
          outB[idx] = (__bf16)y;
        } else {                               // plain f32
          outF[idx] = v;
        }
      }
}

// ---------------------------------------------------------------------------
// Flash attention, causal, + chameleon rotation epilogue.
// Block = 128 threads = 4 waves; wave owns 16 query rows; key tiles of 64.
// Named-scalar register staging + double-buffered K/V; 1 barrier per tile.
// 16 WMMA per wave per tile; every operand load is a b128 pair.
// ---------------------------------------------------------------------------
#define KSTR 72
#define PSTR 72

__global__ __launch_bounds__(128, 1) void attn_wmma(
    const __bf16* __restrict__ qb, const __bf16* __restrict__ kb,
    const __bf16* __restrict__ vt, __bf16* __restrict__ comb,
    const float* __restrict__ cbeta, const float* __restrict__ civ,
    const float* __restrict__ cov,   const float* __restrict__ cchi)
{
  __shared__ __bf16 Ks[2][64 * KSTR];   // [key][dh]
  __shared__ __bf16 Vs[2][64 * KSTR];   // [dh][key]  (from transposed V)
  __shared__ __bf16 Ps[4 * 16 * PSTR];

  int b = blockIdx.z, h = blockIdx.y, qt = blockIdx.x;
  int tid = threadIdx.x, wave = tid >> 5, lane = tid & 31;
  int lh = lane & 15, hi = lane >> 4;
  int q0w = qt * 64 + wave * 16;

  const __bf16* qrow = qb + (size_t)(b * T_ + q0w) * D_ + h * DH_;
  v16bf aq0 = frag_a_vec(qrow,      D_);     // K dims 0..31 of DH
  v16bf aq1 = frag_a_vec(qrow + 32, D_);     // K dims 32..63

  // staging geometry: chunk cc covers rows (tid>>3)+16*cc, cols (tid&7)*8
  int r0 = tid >> 3, q8 = (tid & 7) * 8;
  const __bf16* pK = kb + (size_t)(b * T_ + r0) * D_ + h * DH_ + q8;
  const __bf16* pV = vt + ((size_t)(b * H_ + h) * DH_ + r0) * T_ + q8;
  int ldsO = r0 * KSTR + q8;

  float m[8], l[8];
  v8f o[4] = {};
#pragma unroll
  for (int r = 0; r < 8; ++r) { m[r] = -INFINITY; l[r] = 0.f; }

  int nkt = qt + 1;                          // causal: 64-key tiles
  uint4 tk0, tk1, tk2, tk3, tv0, tv1, tv2, tv3;
  tk0 = *(const uint4*)(pK);
  tk1 = *(const uint4*)(pK + 16 * D_);
  tk2 = *(const uint4*)(pK + 32 * D_);
  tk3 = *(const uint4*)(pK + 48 * D_);
  tv0 = *(const uint4*)(pV);
  tv1 = *(const uint4*)(pV + 16 * T_);
  tv2 = *(const uint4*)(pV + 32 * T_);
  tv3 = *(const uint4*)(pV + 48 * T_);
  pK += (size_t)64 * D_;                     // next tile: 64 keys down
  pV += 64;                                  // next tile: 64 keys right

  int cur = 0;
  for (int kt = 0; kt < nkt; ++kt) {
    *(uint4*)&Ks[cur][ldsO]             = tk0;
    *(uint4*)&Ks[cur][ldsO + 16 * KSTR] = tk1;
    *(uint4*)&Ks[cur][ldsO + 32 * KSTR] = tk2;
    *(uint4*)&Ks[cur][ldsO + 48 * KSTR] = tk3;
    *(uint4*)&Vs[cur][ldsO]             = tv0;
    *(uint4*)&Vs[cur][ldsO + 16 * KSTR] = tv1;
    *(uint4*)&Vs[cur][ldsO + 32 * KSTR] = tv2;
    *(uint4*)&Vs[cur][ldsO + 48 * KSTR] = tv3;
    __syncthreads();

    if (kt + 1 < nkt) {                      // issue next tile's loads now
      tk0 = *(const uint4*)(pK);
      tk1 = *(const uint4*)(pK + 16 * D_);
      tk2 = *(const uint4*)(pK + 32 * D_);
      tk3 = *(const uint4*)(pK + 48 * D_);
      tv0 = *(const uint4*)(pV);
      tv1 = *(const uint4*)(pV + 16 * T_);
      tv2 = *(const uint4*)(pV + 32 * T_);
      tv3 = *(const uint4*)(pV + 48 * T_);
      pK += (size_t)64 * D_;
      pV += 64;
    }

    // scores 16x64 = q (16x64) @ k^T (64x64)
    v8f s[4] = {};
#pragma unroll
    for (int j = 0; j < 4; ++j) {
      s[j] = WMMA_BF16(aq0, frag_b_vec(&Ks[cur][j * 16 * KSTR],      KSTR), s[j]);
      s[j] = WMMA_BF16(aq1, frag_b_vec(&Ks[cur][j * 16 * KSTR + 32], KSTR), s[j]);
    }

#pragma unroll
    for (int r = 0; r < 8; ++r) {
      int qg = q0w + hi * 8 + r;
      float e[4];
#pragma unroll
      for (int j = 0; j < 4; ++j) {
        e[j] = s[j][r] * INV_SCALE_;
        if (kt * 64 + j * 16 + lh > qg) e[j] = -INFINITY;
      }
      float mx = fmaxf(fmaxf(e[0], e[1]), fmaxf(e[2], e[3]));
#pragma unroll
      for (int d = 1; d < 16; d <<= 1) mx = fmaxf(mx, __shfl_xor(mx, d, 32));
      float mn = fmaxf(m[r], mx);            // kt=0 always has key<=q -> mn finite
      float alpha = __expf(m[r] - mn);
      float rs = 0.f;
#pragma unroll
      for (int j = 0; j < 4; ++j) { e[j] = __expf(e[j] - mn); rs += e[j]; }
#pragma unroll
      for (int d = 1; d < 16; d <<= 1) rs += __shfl_xor(rs, d, 32);
      l[r] = l[r] * alpha + rs;
      m[r] = mn;
#pragma unroll
      for (int dj = 0; dj < 4; ++dj) o[dj][r] *= alpha;
#pragma unroll
      for (int j = 0; j < 4; ++j)
        Ps[(wave * 16 + hi * 8 + r) * PSTR + j * 16 + lh] = (__bf16)e[j];
    }
    asm volatile("s_wait_dscnt 0" ::: "memory");   // per-wave LDS RAW fence

    // acc += p (16x64) @ v (64x64)
    v16bf pa0 = frag_a_vec(&Ps[wave * 16 * PSTR],      PSTR);
    v16bf pa1 = frag_a_vec(&Ps[wave * 16 * PSTR + 32], PSTR);
#pragma unroll
    for (int dj = 0; dj < 4; ++dj) {
      o[dj] = WMMA_BF16(pa0, frag_b_vec(&Vs[cur][(dj * 16) * KSTR],      KSTR), o[dj]);
      o[dj] = WMMA_BF16(pa1, frag_b_vec(&Vs[cur][(dj * 16) * KSTR + 32], KSTR), o[dj]);
    }
    cur ^= 1;
  }

  // chameleon epilogue: iv gate, rotation across DH halves, ov*tanh(chi)
  float sb = sigm(cbeta[h]);
  float ang = PI_ * sb, ca = __cosf(ang), sa = __sinf(ang);
  float iv = sigm(civ[h]), g = sigm(cov[h]) * tanhf(cchi[h]);
#pragma unroll
  for (int r = 0; r < 8; ++r) {
    float inv_l = 1.f / l[r];
    float x0 = o[0][r] * inv_l * iv, x1v = o[1][r] * inv_l * iv;
    float x2v = o[2][r] * inv_l * iv, x3 = o[3][r] * inv_l * iv;
    float r0v = (x0 * ca - x2v * sa) * g;
    float r1v = (x1v * ca - x3 * sa) * g;
    float r2v = (x0 * sa + x2v * ca) * g;
    float r3v = (x1v * sa + x3 * ca) * g;
    size_t base = (size_t)(b * T_ + q0w + hi * 8 + r) * D_ + h * DH_ + lh;
    comb[base]      = (__bf16)r0v;
    comb[base + 16] = (__bf16)r1v;
    comb[base + 32] = (__bf16)r2v;
    comb[base + 48] = (__bf16)r3v;
  }
}

// ---------------------------------------------------------------------------
// balance_norm: block per row (D=1024), outputs bf16 for the next GEMM.
// ---------------------------------------------------------------------------
__global__ __launch_bounds__(256) void bnorm_bf16(
    const float* __restrict__ x, const float* __restrict__ g,
    const float* __restrict__ bb, const float* __restrict__ bwP,
    __bf16* __restrict__ out)
{
  __shared__ float red[3][8];
  int row = blockIdx.x, tid = threadIdx.x;
  const float* xr = x + (size_t)row * D_;
  float v[4];
  float s = 0.f, sq = 0.f, sq1 = 0.f;
#pragma unroll
  for (int i = 0; i < 4; ++i) {
    int d = tid + i * 256;
    float t = xr[d];
    v[i] = t; s += t; sq += t * t;
    if (d < D_ / 2) sq1 += t * t;
  }
#pragma unroll
  for (int d2 = 1; d2 < 32; d2 <<= 1) {
    s   += __shfl_xor(s,   d2, 32);
    sq  += __shfl_xor(sq,  d2, 32);
    sq1 += __shfl_xor(sq1, d2, 32);
  }
  int wave = tid >> 5, lane = tid & 31;
  if (lane == 0) { red[0][wave] = s; red[1][wave] = sq; red[2][wave] = sq1; }
  __syncthreads();
  float S = 0.f, SQ = 0.f, SQ1 = 0.f;
#pragma unroll
  for (int w = 0; w < 8; ++w) { S += red[0][w]; SQ += red[1][w]; SQ1 += red[2][w]; }
  float mu  = S / D_;
  float var = SQ / D_ - mu * mu;
  float inv = rsqrtf(var + 1e-6f);
  float e1 = SQ1 / (D_ / 2), e2 = (SQ - SQ1) / (D_ / 2);
  float imb = (e1 - e2) * bwP[0];
#pragma unroll
  for (int i = 0; i < 4; ++i) {
    int d = tid + i * 256;
    float corr = (d < D_ / 2) ? -imb : imb;
    out[(size_t)row * D_ + d] = (__bf16)((v[i] - mu) * inv * g[d] + bb[d] + corr);
  }
}

// ---------------------------------------------------------------------------
// f32 (Rs x Cs, K x N row-major) -> bf16 TRANSPOSED (Np x Kp, N-major),
// zero padded.  LDS-tiled so both global sides are coalesced.
// grid = (Kp/64, Np/64).
// ---------------------------------------------------------------------------
__global__ __launch_bounds__(256) void cvt_t_bf16(
    const float* __restrict__ src, __bf16* __restrict__ dst,
    int Rs, int Cs, int Kp)
{
  __shared__ float Ls[64][65];
  int k0 = blockIdx.x * 64;
  int n0 = blockIdx.y * 64;
  int tid = threadIdx.x;
#pragma unroll
  for (int cc = 0; cc < 16; ++cc) {
    int ch = tid + cc * 256;
    int kk = ch >> 6, nn = ch & 63;
    int gk = k0 + kk, gn = n0 + nn;
    Ls[kk][nn] = (gk < Rs && gn < Cs) ? src[(size_t)gk * Cs + gn] : 0.f;
  }
  __syncthreads();
#pragma unroll
  for (int cc = 0; cc < 16; ++cc) {
    int ch = tid + cc * 256;
    int nn = ch >> 6, kk = ch & 63;
    dst[(size_t)(n0 + nn) * Kp + k0 + kk] = (__bf16)Ls[kk][nn];
  }
}

// ---------------------------------------------------------------------------
// V transpose: [B,T,H*DH] bf16 -> [B,H,DH,T] bf16 (one-time, LDS-tiled).
// grid = (T/64, H, B).
// ---------------------------------------------------------------------------
__global__ __launch_bounds__(256) void transpose_v(
    const __bf16* __restrict__ vbuf, __bf16* __restrict__ vt)
{
  __shared__ __bf16 Ls[64 * 72];
  int t0 = blockIdx.x * 64, h = blockIdx.y, b = blockIdx.z;
  int tid = threadIdx.x;
#pragma unroll
  for (int cc = 0; cc < 2; ++cc) {
    int ch = tid + cc * 256;
    int t = ch >> 3, q = ch & 7;
    *(uint4*)&Ls[t * 72 + q * 8] =
        *(const uint4*)&vbuf[(size_t)(b * T_ + t0 + t) * D_ + h * DH_ + q * 8];
  }
  __syncthreads();
#pragma unroll
  for (int cc = 0; cc < 2; ++cc) {
    int ch = tid + cc * 256;
    int dh = ch >> 3, q = ch & 7;
    union { uint4 u; __bf16 hh[8]; } U;
#pragma unroll
    for (int w = 0; w < 8; ++w) U.hh[w] = Ls[(q * 8 + w) * 72 + dh];
    *(uint4*)&vt[((size_t)(b * H_ + h) * DH_ + dh) * T_ + t0 + q * 8] = U.u;
  }
}

// ---------------------------------------------------------------------------
extern "C" void kernel_launch(void* const* d_in, const int* in_sizes, int n_in,
                              void* d_out, int out_size, void* d_ws, size_t ws_size,
                              hipStream_t stream)
{
  (void)in_sizes; (void)n_in; (void)out_size; (void)ws_size;
  const float* x     = (const float*)d_in[0];
  // d_in[1] = mask (causal tril) -- implemented analytically
  const float* Wq    = (const float*)d_in[2];
  const float* Wk    = (const float*)d_in[3];
  const float* Wv    = (const float*)d_in[4];
  const float* Wo    = (const float*)d_in[5];
  const float* ln1g  = (const float*)d_in[6];
  const float* ln1b  = (const float*)d_in[7];
  const float* bw1   = (const float*)d_in[8];
  const float* ln2g  = (const float*)d_in[9];
  const float* ln2b  = (const float*)d_in[10];
  const float* bw2   = (const float*)d_in[11];
  const float* cbeta = (const float*)d_in[12];
  const float* civ   = (const float*)d_in[13];
  const float* cov   = (const float*)d_in[14];
  const float* cchi  = (const float*)d_in[15];
  const float* Win   = (const float*)d_in[16];
  const float* bin   = (const float*)d_in[17];
  const float* Wout  = (const float*)d_in[18];
  const float* bout  = (const float*)d_in[19];
  const float* rbeta = (const float*)d_in[20];
  const float* nestW = (const float*)d_in[21];
  float* out = (float*)d_out;

  char* p = (char*)d_ws;
  auto alloc = [&](size_t bytes) -> void* {
    void* r = (void*)p; p += (bytes + 255) & ~(size_t)255; return r;
  };
  // transposed bf16 weights: Wt[n][k]
  __bf16* Wq_t   = (__bf16*)alloc((size_t)D_ * D_ * 2);
  __bf16* Wk_t   = (__bf16*)alloc((size_t)D_ * D_ * 2);
  __bf16* Wv_t   = (__bf16*)alloc((size_t)D_ * D_ * 2);
  __bf16* Wo_t   = (__bf16*)alloc((size_t)D_ * D_ * 2);
  __bf16* nest_t = (__bf16*)alloc((size_t)D_ * D_ * 2);
  __bf16* Win_t  = (__bf16*)alloc((size_t)DHIDP_ * D_ * 2);   // 2176 x 1024
  __bf16* Wout_t = (__bf16*)alloc((size_t)D_ * DHIDP_ * 2);   // 1024 x 2176
  __bf16* xn1b   = (__bf16*)alloc((size_t)BT_ * D_ * 2);
  __bf16* qbuf   = (__bf16*)alloc((size_t)BT_ * D_ * 2);
  __bf16* kbuf   = (__bf16*)alloc((size_t)BT_ * D_ * 2);
  __bf16* vbuf   = (__bf16*)alloc((size_t)BT_ * D_ * 2);
  __bf16* vtbuf  = (__bf16*)alloc((size_t)BT_ * D_ * 2);      // [B,H,DH,T]
  __bf16* combb  = (__bf16*)alloc((size_t)BT_ * D_ * 2);
  __bf16* xn2b   = (__bf16*)alloc((size_t)BT_ * D_ * 2);
  __bf16* hidb   = (__bf16*)alloc((size_t)BT_ * DHIDP_ * 2);
  __bf16* x2b    = (__bf16*)alloc((size_t)BT_ * D_ * 2);
  float*  x1     = (float*) alloc((size_t)BT_ * D_ * 4);
  float*  x2     = (float*) alloc((size_t)BT_ * D_ * 4);

  // weight transpose+convert: src K x N -> dst Np x Kp (N-major), zero pad
  auto cvtT = [&](const float* s, __bf16* dst, int Ksrc, int Nsrc, int Kp, int Np) {
    cvt_t_bf16<<<dim3(Kp / 64, Np / 64), 256, 0, stream>>>(s, dst, Ksrc, Nsrc, Kp);
  };
  cvtT(Wq,    Wq_t,   D_, D_,       D_,     D_);
  cvtT(Wk,    Wk_t,   D_, D_,       D_,     D_);
  cvtT(Wv,    Wv_t,   D_, D_,       D_,     D_);
  cvtT(Wo,    Wo_t,   D_, D_,       D_,     D_);
  cvtT(nestW, nest_t, D_, D_,       D_,     D_);
  cvtT(Win,   Win_t,  D_, DHID_,    D_,     DHIDP_);
  cvtT(Wout,  Wout_t, DHID_, D_,    DHIDP_, D_);

  dim3 gD (D_ / GBN,     BT_ / GBM);   // N=1024 -> 16 x 32
  dim3 gH (DHIDP_ / GBN, BT_ / GBM);   // N=2176 -> 34 x 32

  // ln1 -> xn1
  bnorm_bf16<<<BT_, 256, 0, stream>>>(x, ln1g, ln1b, bw1, xn1b);
  // Q, K, V projections (mode 0: bf16 store)
  gemm_bf16_wmma<<<gD, 128, 0, stream>>>(xn1b, Wq_t, BT_, D_, D_, qbuf, nullptr, nullptr, 0, nullptr, nullptr, 0, 0);
  gemm_bf16_wmma<<<gD, 128, 0, stream>>>(xn1b, Wk_t, BT_, D_, D_, kbuf, nullptr, nullptr, 0, nullptr, nullptr, 0, 0);
  gemm_bf16_wmma<<<gD, 128, 0, stream>>>(xn1b, Wv_t, BT_, D_, D_, vbuf, nullptr, nullptr, 0, nullptr, nullptr, 0, 0);
  // V -> [B,H,DH,T] for dh-major attention tiles
  transpose_v<<<dim3(T_ / 64, H_, B_), 256, 0, stream>>>(vbuf, vtbuf);
  // flash attention + chameleon
  attn_wmma<<<dim3(T_ / 64, H_, B_), 128, 0, stream>>>(qbuf, kbuf, vtbuf, combb, cbeta, civ, cov, cchi);
  // Wo projection + residual (mode 1): x1 = x + rb * (comb @ Wo)
  gemm_bf16_wmma<<<gD, 128, 0, stream>>>(combb, Wo_t, BT_, D_, D_, nullptr, x1, nullptr, 0, x, rbeta, 0, 1);
  // ln2 -> xn2
  bnorm_bf16<<<BT_, 256, 0, stream>>>(x1, ln2g, ln2b, bw2, xn2b);
  // FFN in (mode 2): hidden = gelu(mask(xn2 @ Win + b_in))
  gemm_bf16_wmma<<<gH, 128, 0, stream>>>(xn2b, Win_t, BT_, DHIDP_, D_, hidb, nullptr, bin, DHID_, nullptr, nullptr, DACT_, 2);
  // FFN out + residual (mode 3): x2 = x1 + rb * (hidden @ Wout + b_out)
  gemm_bf16_wmma<<<gD, 128, 0, stream>>>(hidb, Wout_t, BT_, D_, DHIDP_, x2b, x2, bout, D_, x1, rbeta, 0, 3);
  // nest_W (mode 4): out = x2 @ nest_W
  gemm_bf16_wmma<<<gD, 128, 0, stream>>>(x2b, nest_t, BT_, D_, D_, nullptr, out, nullptr, 0, nullptr, nullptr, 0, 4);
}
